// GraphVAE_66589172957277
// MI455X (gfx1250) — compile-verified, matrix operand order checked by reading, not run on hardware
//
#include <hip/hip_runtime.h>
#include <hip/hip_bf16.h>

// ---------------------------------------------------------------------------
// GraphVAE (2-layer GCN encoder + reparameterization) for MI455X / gfx1250.
// GEMMs run on v_wmma_f32_16x16x32_bf16 (fp32 accumulate, bf16 inputs).
// 64-row M-tiles: each wave carries 4 accumulators and reuses its B fragment
// across 4 WMMAs per K-slab (amortizes LDS staging + barriers 4x).
// Weights are pre-transposed so both operands load from LDS as ds_load_b128.
// Edge aggregation uses global f32 atomics (feature matrices fit in 192MB L2).
// ---------------------------------------------------------------------------

typedef __attribute__((ext_vector_type(16))) __bf16 v16bf;
typedef __attribute__((ext_vector_type(8)))  float  v8f;

#define F_OUT 128          // output feature width (fixed by reference)
#define M_TILE 64          // rows of A per block (4 WMMA sub-tiles per wave)
#define MAX_LOGSTD 10.0f
#define SELF_LOOP_W 2.0f

// ---------------- elementwise helpers ----------------

// src[K][N] row-major f32  ->  dst[N][K] bf16 (transposed)
__global__ void k_f32_to_bf16_T(const float* __restrict__ src,
                                __bf16* __restrict__ dst, int K, int N) {
  int i = blockIdx.x * blockDim.x + threadIdx.x;
  if (i < K * N) {
    int k = i / N, n = i - k * N;
    dst[(size_t)n * K + k] = (__bf16)src[i];
  }
}

__global__ void k_init_deg(float* __restrict__ deg, int n) {
  int i = blockIdx.x * blockDim.x + threadIdx.x;
  if (i < n) deg[i] = SELF_LOOP_W;   // self loop contributes w=2.0 at each node
}

__global__ void k_edge_deg(float* __restrict__ deg,
                           const long long* __restrict__ ei,
                           const float* __restrict__ w, int E) {
  int e = blockIdx.x * blockDim.x + threadIdx.x;
  if (e < E) atomicAdd(&deg[ei[(size_t)E + e]], w[e]);   // col = ei[1][e]
}

__global__ void k_dinv(const float* __restrict__ deg,
                       float* __restrict__ dinv, int n) {
  int i = blockIdx.x * blockDim.x + threadIdx.x;
  if (i < n) {
    float d = deg[i];
    dinv[i] = d > 0.0f ? __frsqrt_rn(d) : 0.0f;
  }
}

// ---------------- WMMA GEMM: C[M,128] = A[M,K] * B^T[128,K] ----------------
// Block = 256 threads (8 waves). Block owns 64 rows of A; wave w computes the
// 64x16 C strip at columns [16w, 16w+16) as 4 stacked 16x16 WMMA tiles.
// A is staged through LDS with on-the-fly f32->bf16 conversion (AT=float for
// x, AT=__bf16 for h). B is supplied PRE-TRANSPOSED (BT[n][k]) so its LDS
// stage and fragment reads mirror A's (all ds_load_b128).
//
// CDNA5 16-bit A layout (16x32): lane m (0-15) K in {0..7,16..23},
// lane m+16 K in {8..15,24..31}; element j -> K = 8*hi + j + (j&8).
// B (32x16) mirrors with N across lanes (same per-lane K pattern).
// C/D: VGPR r holds M = r + 8*hi, N = lane&15.

template <typename AT>
__global__ __launch_bounds__(256)
void k_wmma_gemm(const AT* __restrict__ A, const __bf16* __restrict__ BT,
                 float* __restrict__ C, int M, int K) {
  __shared__ __bf16 As[M_TILE][40];     // 64 (M) x 32 (K) tile (+pad, 16B-aligned rows)
  __shared__ __bf16 BsT[F_OUT][40];     // 128 (N) x 32 (K) tile (+pad)

  const int tid    = threadIdx.x;
  const int wave   = tid >> 5;
  const int lane   = tid & 31;
  const int m_base = blockIdx.x * M_TILE;
  const int n_base = wave * 16;
  const int mm     = lane & 15;
  const int hi     = lane >> 4;

  // Hoisted per-thread staging pointers (row index is kb-invariant).
  const int a_r = tid >> 2;                 // A stage: row 0..63, 8 elems each
  const int a_c = (tid & 3) * 8;
  int a_row = m_base + a_r;
  if (a_row >= M) a_row = M - 1;            // clamp once for the ragged block
  const AT* a_src = A + (size_t)a_row * K + a_c;
  __bf16*   a_dst = &As[a_r][a_c];

  const int b_r  = tid >> 1;                // B stage: n row 0..127, 16 each
  const int b_c0 = (tid & 1) * 16;
  const __bf16* b_src = BT + (size_t)b_r * K + b_c0;
  __bf16*       b_dst = &BsT[b_r][b_c0];

  v8f acc[4] = {};

  for (int kb = 0; kb < K; kb += 32) {
    __syncthreads();
    { // stage A: 64x32 = 2048 elements, 8 contiguous per thread (+cvt)
      const AT* ap = a_src + kb;
#pragma unroll
      for (int i = 0; i < 8; ++i) a_dst[i] = (__bf16)ap[i];
    }
    { // stage B^T: 128x32 = 4096 bf16, 16 contiguous per thread (2x b128)
      const __bf16* bp = b_src + kb;
#pragma unroll
      for (int i = 0; i < 16; ++i) b_dst[i] = bp[i];
    }
    __syncthreads();

    v16bf b;
#pragma unroll
    for (int j = 0; j < 16; ++j) {
      int k = 8 * hi + j + (j & 8);     // contiguous runs of 8 -> ds_load_b128 x2
      b[j] = BsT[n_base + mm][k];
    }
#pragma unroll
    for (int s = 0; s < 4; ++s) {       // 4 stacked M sub-tiles share b
      v16bf a;
#pragma unroll
      for (int j = 0; j < 16; ++j) {
        int k = 8 * hi + j + (j & 8);
        a[j] = As[s * 16 + mm][k];
      }
      acc[s] = __builtin_amdgcn_wmma_f32_16x16x32_bf16(
          false, a, false, b, (short)0, acc[s], false, false);
    }
  }

  // C epilogue: uniform branch keeps the common path free of exec-mask ops.
  float* cp = C + (size_t)m_base * F_OUT + (size_t)(8 * hi) * F_OUT + n_base + mm;
  if (m_base + M_TILE <= M) {           // fast path: full tile (781/782 blocks)
#pragma unroll
    for (int s = 0; s < 4; ++s) {
#pragma unroll
      for (int r = 0; r < 8; ++r)
        cp[(size_t)(s * 16 + r) * F_OUT] = acc[s][r];
    }
  } else {                              // ragged tail: per-row guard
#pragma unroll
    for (int s = 0; s < 4; ++s) {
#pragma unroll
      for (int r = 0; r < 8; ++r) {
        int m = m_base + s * 16 + r + 8 * hi;
        if (m < M) C[(size_t)m * F_OUT + n_base + mm] = acc[s][r];
      }
    }
  }
}

// ---------------- edge scatter: dst[col] += src[row] * norm ----------------
// One wave per edge; each lane moves float4 (4*32 = 128 features).

__global__ __launch_bounds__(256)
void k_scatter(const float* __restrict__ src, float* __restrict__ dst,
               const long long* __restrict__ ei, const float* __restrict__ w,
               const float* __restrict__ dinv, int E) {
  int gid  = blockIdx.x * blockDim.x + threadIdx.x;
  int e    = gid >> 5;
  int lane = gid & 31;
  if (e >= E) return;
  long long r = ei[e];
  long long c = ei[(size_t)E + e];
  float nrm = dinv[r] * w[e] * dinv[c];
  const float4 v = *(const float4*)(src + (size_t)r * F_OUT + lane * 4);
  float* d = dst + (size_t)c * F_OUT + lane * 4;
  atomicAdd(d + 0, v.x * nrm);
  atomicAdd(d + 1, v.y * nrm);
  atomicAdd(d + 2, v.z * nrm);
  atomicAdd(d + 3, v.w * nrm);
}

// ---------------- conv1 epilogue: self-loop + bias + ReLU -> bf16 ----------

__global__ void k_finalize1(const float* __restrict__ agg,
                            const float* __restrict__ xw,
                            const float* __restrict__ dinv,
                            const float* __restrict__ b1,
                            __bf16* __restrict__ h_bf, int total) {
  int i = blockIdx.x * blockDim.x + threadIdx.x;
  if (i >= total) return;
  int node = i >> 7, f = i & (F_OUT - 1);
  float di = dinv[node];
  float sn = SELF_LOOP_W * di * di;              // self-loop norm
  float h = agg[i] + xw[i] * sn + b1[f];
  h_bf[i] = (__bf16)fmaxf(h, 0.0f);
}

// ---------------- final: z = mu + eps * exp(min(logstd, 10)) ---------------

__global__ void k_final(const float* __restrict__ agg_mu,
                        const float* __restrict__ hw2,
                        const float* __restrict__ agg_ls,
                        const float* __restrict__ hw3,
                        const float* __restrict__ dinv,
                        const float* __restrict__ b2,
                        const float* __restrict__ b3,
                        const float* __restrict__ eps,
                        float* __restrict__ z, int total) {
  int i = blockIdx.x * blockDim.x + threadIdx.x;
  if (i >= total) return;
  int node = i >> 7, f = i & (F_OUT - 1);
  float di = dinv[node];
  float sn = SELF_LOOP_W * di * di;
  float mu = agg_mu[i] + hw2[i] * sn + b2[f];
  float ls = fminf(agg_ls[i] + hw3[i] * sn + b3[f], MAX_LOGSTD);
  z[i] = mu + eps[i] * __expf(ls);
}

// ---------------------------------------------------------------------------

static inline size_t align256(size_t x) { return (x + 255) & ~(size_t)255; }

extern "C" void kernel_launch(void* const* d_in, const int* in_sizes, int n_in,
                              void* d_out, int out_size, void* d_ws, size_t ws_size,
                              hipStream_t stream) {
  const float*     x   = (const float*)d_in[0];
  const long long* ei  = (const long long*)d_in[1];   // int64 [2,E]
  const float*     ew  = (const float*)d_in[2];
  const float*     eps = (const float*)d_in[3];
  const float*     W1  = (const float*)d_in[4];
  const float*     b1  = (const float*)d_in[5];
  const float*     W2  = (const float*)d_in[6];
  const float*     b2  = (const float*)d_in[7];
  const float*     W3  = (const float*)d_in[8];
  const float*     b3  = (const float*)d_in[9];

  const int Fo = in_sizes[5];            // 128
  const int Fi = in_sizes[4] / Fo;       // 512
  const int Nn = in_sizes[0] / Fi;       // 50000 nodes
  const int E  = in_sizes[1] / 2;        // 800000 edges
  const int NF = Nn * Fo;                // node-feature element count

  // ---- workspace bump allocator ----
  char* ws = (char*)d_ws;
  size_t off = 0;
  auto alloc = [&](size_t bytes) -> void* {
    void* p = ws + off;
    off = align256(off + bytes);
    return p;
  };
  float*  deg   = (float*)alloc((size_t)Nn * 4);
  float*  dinv  = (float*)alloc((size_t)Nn * 4);
  __bf16* W1t   = (__bf16*)alloc((size_t)Fi * Fo * 2);   // [Fo][Fi] transposed
  __bf16* W2t   = (__bf16*)alloc((size_t)Fo * Fo * 2);   // [Fo][Fo] transposed
  __bf16* W3t   = (__bf16*)alloc((size_t)Fo * Fo * 2);   // [Fo][Fo] transposed
  float*  xw    = (float*)alloc((size_t)NF * 4);   // reused as hw2
  float*  agg1  = (float*)alloc((size_t)NF * 4);   // reused as agg_mu
  __bf16* h_bf  = (__bf16*)alloc((size_t)NF * 2);
  float*  hw3   = (float*)alloc((size_t)NF * 4);
  float*  aggls = (float*)alloc((size_t)NF * 4);
  (void)ws_size; (void)n_in; (void)out_size;

  const int T = 256;
  auto nblk = [](long long n, int t) { return (int)((n + t - 1) / t); };

  // weights -> bf16, transposed to [N][K] so GEMM B-staging is coalesced
  k_f32_to_bf16_T<<<nblk((long long)Fi * Fo, T), T, 0, stream>>>(W1, W1t, Fi, Fo);
  k_f32_to_bf16_T<<<nblk((long long)Fo * Fo, T), T, 0, stream>>>(W2, W2t, Fo, Fo);
  k_f32_to_bf16_T<<<nblk((long long)Fo * Fo, T), T, 0, stream>>>(W3, W3t, Fo, Fo);

  // degree (with self loops) and rsqrt norm
  k_init_deg<<<nblk(Nn, T), T, 0, stream>>>(deg, Nn);
  k_edge_deg<<<nblk(E, T), T, 0, stream>>>(deg, ei, ew, E);
  k_dinv<<<nblk(Nn, T), T, 0, stream>>>(deg, dinv, Nn);

  const int gemm_blocks = (Nn + M_TILE - 1) / M_TILE;    // 782
  const int scat_blocks = nblk((long long)E * 32, T);

  // ---- conv1: h = relu(norm-agg(x@W1) + b1) ----
  hipMemsetAsync(agg1, 0, (size_t)NF * 4, stream);
  k_wmma_gemm<float><<<gemm_blocks, T, 0, stream>>>(x, W1t, xw, Nn, Fi);
  k_scatter<<<scat_blocks, T, 0, stream>>>(xw, agg1, ei, ew, dinv, E);
  k_finalize1<<<nblk(NF, T), T, 0, stream>>>(agg1, xw, dinv, b1, h_bf, NF);

  // ---- conv2 (mu) and conv3 (logstd); reuse xw->hw2, agg1->agg_mu ----
  float* hw2    = xw;
  float* agg_mu = agg1;
  k_wmma_gemm<__bf16><<<gemm_blocks, T, 0, stream>>>(h_bf, W2t, hw2, Nn, Fo);
  k_wmma_gemm<__bf16><<<gemm_blocks, T, 0, stream>>>(h_bf, W3t, hw3, Nn, Fo);

  hipMemsetAsync(agg_mu, 0, (size_t)NF * 4, stream);
  hipMemsetAsync(aggls, 0, (size_t)NF * 4, stream);
  k_scatter<<<scat_blocks, T, 0, stream>>>(hw2, agg_mu, ei, ew, dinv, E);
  k_scatter<<<scat_blocks, T, 0, stream>>>(hw3, aggls, ei, ew, dinv, E);

  // ---- reparameterization ----
  k_final<<<nblk(NF, T), T, 0, stream>>>(agg_mu, hw2, aggls, hw3, dinv, b2, b3,
                                         eps, (float*)d_out, NF);
}
